// GNNObservationEncoder_23691039605127
// MI455X (gfx1250) — compile-verified, matrix operand order checked by reading, not run on hardware
//
#include <hip/hip_runtime.h>
#include <stdint.h>

#define LEAKY_ALPHA 0.2f
#define MASK_VAL   -9.0e15f

constexpr int Bc = 32, Nc = 1024, Fc = 64, Hc = 3, Dc = 16;

typedef __attribute__((ext_vector_type(16))) _Float16 v16h;
typedef __attribute__((ext_vector_type(8)))  _Float16 v8h;
typedef __attribute__((ext_vector_type(2)))  _Float16 h2;
typedef __attribute__((ext_vector_type(8)))  float    v8f;
typedef __attribute__((ext_vector_type(4)))  float    f4;
typedef __attribute__((ext_vector_type(4)))  int      v4i;

#if defined(__has_builtin)
#if __has_builtin(__builtin_amdgcn_global_load_async_to_lds_b128)
#define USE_ASYNC_LDS 1
#endif
#endif

// K index inside a 16x16x32 f16 WMMA fragment for (vgpr v, half p, lane-half hi):
// v0..3 -> K 0..7 (lo lanes) / 8..15 (hi lanes); v4..7 -> +16. (ISA 7.12.2)
__device__ __forceinline__ int kmap(int v, int p, int hi) {
  return ((v & 4) << 2) + (hi << 3) + ((v & 3) << 1) + p;
}

// Copy n16*16 bytes global -> LDS. Prefers gfx1250 async-to-LDS (ASYNCcnt,
// no VGPR round-trip); falls back to b128 vector copies.
__device__ __forceinline__ void copy_to_lds_b128(void* lds_dst, const void* gsrc,
                                                 int n16, int tid, int nthreads) {
#if defined(USE_ASYNC_LDS)
  typedef __attribute__((address_space(1))) v4i g_v4i;
  typedef __attribute__((address_space(3))) v4i l_v4i;
  g_v4i* gp = (g_v4i*)(uintptr_t)gsrc;               // drop const: builtin takes non-const
  l_v4i* lp = (l_v4i*)(uint32_t)(uintptr_t)lds_dst;  // low 32 bits = LDS offset
  for (int i = tid; i < n16; i += nthreads)
    __builtin_amdgcn_global_load_async_to_lds_b128(gp + i, lp + i, 0, 0);
#else
  const f4* src = (const f4*)gsrc;
  f4* dst = (f4*)lds_dst;
  for (int i = tid; i < n16; i += nthreads) dst[i] = src[i];
#endif
}

__device__ __forceinline__ void wait_lds_copies() {
#if defined(USE_ASYNC_LDS)
#if __has_builtin(__builtin_amdgcn_s_wait_asynccnt)
  __builtin_amdgcn_s_wait_asynccnt(0);
#else
  asm volatile("s_wait_asynccnt 0x0" ::: "memory");
#endif
#endif
}

// ---------------------------------------------------------------------------
// Kernel 0: pack adj>0 into a bitmask, once for all 96 (b,h) pairs.
// adj re-reads were ~400 MB (95% of HBM traffic); the 128 KB mask is L2-hot.
// One wave packs 32 consecutive floats (128 B coalesced) -> 1 u32 via ballot.
// ---------------------------------------------------------------------------
__global__ void __launch_bounds__(256)
adj_pack_kernel(const float* __restrict__ adj, uint32_t* __restrict__ mask) {
  const int lane = threadIdx.x & 31;
  const int wavegid = (int)((blockIdx.x * 256 + threadIdx.x) >> 5);
  const int nwaves = (512 * 256) / 32;
  for (int w = wavegid; w < (Nc * Nc) / 32; w += nwaves) {
    const float v = adj[(size_t)w * 32 + lane];
    const unsigned long long b = __ballot(v > 0.f);
    if (lane == 0) mask[w] = (uint32_t)b;
  }
}

// ---------------------------------------------------------------------------
// Kernel 1: Wh = h @ W  (per b,h), plus s1 = Wh.a1, s2 = Wh.a2.
// One wave per 16-row tile; 2x v_wmma_f32_16x16x32_f16 (K=64).
// Stores Wh as f16 transposed (B,H,D,N) for LDS-friendly reuse in kernel 2.
// ---------------------------------------------------------------------------
__global__ void __launch_bounds__(32)
gat_wh_kernel(const float* __restrict__ hsrc,   // (B,N,F)
              const float* __restrict__ W,      // (H,F,D)
              const float* __restrict__ a,      // (H,2D,1)
              _Float16* __restrict__ wh16t,     // (B,H,D,N) f16
              float* __restrict__ s1,           // (B,H,N)
              float* __restrict__ s2)           // (B,H,N)
{
  __shared__ float tile[16][16];

  const int wg = blockIdx.x;
  const int it = wg & 63;            // N/16 = 64 tiles
  const int hh = (wg >> 6) % Hc;
  const int b  = wg / (64 * Hc);
  const int lane = threadIdx.x;
  const int m  = lane & 15;
  const int hi = lane >> 4;

  const float* hrow = hsrc + (size_t)(b * Nc + it * 16 + m) * Fc;
  const float* Wmat = W + hh * Fc * Dc;

  v8f c = {};
#pragma unroll
  for (int ks = 0; ks < 2; ++ks) {
    v16h af, bf;
#pragma unroll
    for (int v = 0; v < 8; ++v) {
#pragma unroll
      for (int p = 0; p < 2; ++p) {
        const int k = ks * 32 + kmap(v, p, hi);
        af[2 * v + p] = (_Float16)hrow[k];            // A: row m, col k
        bf[2 * v + p] = (_Float16)Wmat[k * Dc + m];   // B: row k, col m
      }
    }
    c = __builtin_amdgcn_wmma_f32_16x16x32_f16(false, af, false, bf,
                                               (short)0, c, false, false);
  }

  // Stage C tile (row = v + 8*hi, col = m) for the attention-score dots.
#pragma unroll
  for (int v = 0; v < 8; ++v) tile[v + 8 * hi][m] = c[v];
  __syncthreads();

  // lanes 0-15 compute s1[row=m]; lanes 16-31 compute s2[row=m]
  {
    const float* av = a + hh * 2 * Dc + hi * Dc;
    float s = 0.f;
#pragma unroll
    for (int d = 0; d < Dc; ++d) s += tile[m][d] * av[d];
    float* dst = hi ? s2 : s1;
    dst[(size_t)(b * Hc + hh) * Nc + it * 16 + m] = s;
  }

  // Store Wh tile as f16, transposed: wh16t[b,h,d=m, n = it*16 + v + 8*hi]
  {
    v8h o;
#pragma unroll
    for (int v = 0; v < 8; ++v) o[v] = (_Float16)c[v];
    _Float16* dst = wh16t + (size_t)((b * Hc + hh) * Dc + m) * Nc
                         + it * 16 + 8 * hi;           // 16B aligned
    *reinterpret_cast<v8h*>(dst) = o;
  }
}

// ---------------------------------------------------------------------------
// Kernel 2: flash-attention style softmax(e) @ Wh.
// 4 waves/block share LDS copies of s2 (4KB) and Wh^T f16 (32KB) for one
// (b,h); each wave owns one 16-row i-tile: 32 chunked
// v_wmma_f32_16x16x32_f16 accumulations with online softmax in registers.
// adj handled via 1 u32 bitmask word per lane per chunk.
// ---------------------------------------------------------------------------
__global__ void __launch_bounds__(128)
gat_attn_kernel(const uint32_t* __restrict__ mask,    // (N, N/32) bits
                const _Float16* __restrict__ wh16t,   // (B,H,D,N)
                const float* __restrict__ s1,         // (B,H,N)
                const float* __restrict__ s2,         // (B,H,N)
                float* __restrict__ out)              // (B,N,H*D)
{
  __shared__ __align__(16) float    s2s[Nc];
  __shared__ __align__(16) _Float16 whts[Dc * Nc];

  const int wg  = blockIdx.x;
  const int itb = wg & 15;           // 16 blocks x 4 tiles = 64 tiles
  const int hh  = (wg >> 4) % Hc;
  const int b   = wg / (16 * Hc);
  const int bh  = b * Hc + hh;

  const int tid  = threadIdx.x;
  const int wave = tid >> 5;
  const int lane = tid & 31;
  const int m    = lane & 15;
  const int hi   = lane >> 4;
  const int it   = itb * 4 + wave;

  // Cooperative preload (shared across the 4 waves of this block).
  copy_to_lds_b128(s2s,  s2 + (size_t)bh * Nc,         Nc * 4 / 16,      tid, 128);
  copy_to_lds_b128(whts, wh16t + (size_t)bh * Dc * Nc, Dc * Nc * 2 / 16, tid, 128);
  wait_lds_copies();
  __syncthreads();

  const int   iglob = it * 16 + m;                 // stat row for this lane
  const float s1v   = s1[(size_t)bh * Nc + iglob];
  const uint32_t* maskrow = mask + (size_t)iglob * (Nc / 32);
  const h2* wcol = reinterpret_cast<const h2*>(whts + m * Nc);  // col = m

  float rowM = -__builtin_inff();
  float rowS = 0.f;
  v8f   acc  = {};

  for (int jc = 0; jc < Nc; jc += 32) {
    // Mask word for this row/chunk, pre-shifted so bit tests are constants.
    const uint32_t mws = maskrow[jc >> 5] >> (hi << 3);

    // Masked leaky-relu logits, built directly in A-fragment order.
    float e[16];
    float cmax = -__builtin_inff();
#pragma unroll
    for (int v = 0; v < 8; ++v) {
#pragma unroll
      for (int p = 0; p < 2; ++p) {
        const int j = jc + kmap(v, p, hi);
        float t = s1v + s2s[j];
        t = fmaxf(t, LEAKY_ALPHA * t);               // leaky relu, alpha<1
        const uint32_t bit = 1u << (((v & 4) << 2) + ((v & 3) << 1) + p);
        t = (mws & bit) ? t : MASK_VAL;
        e[2 * v + p] = t;
        cmax = fmaxf(cmax, t);
      }
    }
    // Online softmax: partner lane (xor 16) holds the other half of row m.
    cmax = fmaxf(cmax, __shfl_xor(cmax, 16, 32));
    const float newM = fmaxf(rowM, cmax);
    const float corr = __expf(rowM - newM);          // 0 on first chunk
    rowM = newM;

    v16h af;
    float psum = 0.f;
#pragma unroll
    for (int q = 0; q < 16; ++q) {
      const float pv = __expf(e[q] - rowM);
      psum += pv;
      af[q] = (_Float16)pv;
    }
    psum += __shfl_xor(psum, 16, 32);
    rowS = rowS * corr + psum;

    // B fragment: Wh^T rows are contiguous in LDS -> paired u32 reads.
    v16h bf;
#pragma unroll
    for (int v = 0; v < 8; ++v) {
      const h2 pr = wcol[(jc + kmap(v, 0, hi)) >> 1];
      bf[2 * v]     = pr.x;
      bf[2 * v + 1] = pr.y;
    }

    // Rescale accumulator rows (C row = v + 8*hi; stats live in lane == row).
#pragma unroll
    for (int v = 0; v < 8; ++v)
      acc[v] *= __shfl(corr, v + 8 * hi, 32);

    acc = __builtin_amdgcn_wmma_f32_16x16x32_f16(false, af, false, bf,
                                                 (short)0, acc, false, false);
  }

  // Normalize and write with fused transpose to (B, N, H*D).
#pragma unroll
  for (int v = 0; v < 8; ++v) {
    const float S = __shfl(rowS, v + 8 * hi, 32);
    const int n = it * 16 + v + 8 * hi;
    out[(size_t)(b * Nc + n) * (Hc * Dc) + hh * Dc + m] = acc[v] / S;
  }
}

// ---------------------------------------------------------------------------
extern "C" void kernel_launch(void* const* d_in, const int* in_sizes, int n_in,
                              void* d_out, int out_size, void* d_ws, size_t ws_size,
                              hipStream_t stream) {
  const float* h   = (const float*)d_in[0];   // (32,1024,64)
  const float* adj = (const float*)d_in[1];   // (1024,1024)
  const float* W   = (const float*)d_in[2];   // (3,64,16)
  const float* a   = (const float*)d_in[3];   // (3,32,1)
  float* out = (float*)d_out;                 // (32,1024,48)

  char* ws = (char*)d_ws;
  const size_t wh_bytes   = (size_t)Bc * Hc * Dc * Nc * sizeof(_Float16); // 3 MB
  const size_t s_bytes    = (size_t)Bc * Hc * Nc * sizeof(float);        // 384 KB
  _Float16* wh16t = (_Float16*)ws;
  float*    s1    = (float*)(ws + wh_bytes);
  float*    s2    = (float*)(ws + wh_bytes + s_bytes);
  uint32_t* mask  = (uint32_t*)(ws + wh_bytes + 2 * s_bytes);            // 128 KB

  // Kernel 0: pack adjacency once (shared by all b,h).
  adj_pack_kernel<<<dim3(512), dim3(256), 0, stream>>>(adj, mask);

  // Kernel 1: 32*3*64 tiles, one wave each.
  gat_wh_kernel<<<dim3(Bc * Hc * 64), dim3(32), 0, stream>>>(
      h, W, a, wh16t, s1, s2);

  // Kernel 2: 32*3*16 blocks, 4 waves each (one i-tile per wave).
  gat_attn_kernel<<<dim3(Bc * Hc * 16), dim3(128), 0, stream>>>(
      mask, wh16t, s1, s2, out);
}
